// WindowAttention_28149215658498
// MI455X (gfx1250) — matrix operand order repaired and checked by reading, not compiled, not run on hardware
//
#include <hip/hip_runtime.h>
#include <hip/hip_bf16.h>

// ---------------- problem constants ----------------
#define BATCH   32
#define NHW     64          // NH == NW
#define DIMC    384
#define HEADS   12
#define HDIM    32
#define WS      8
#define M2      64          // WS*WS
#define NWIN    64          // (64/8)^2
#define INNER   384         // HEADS*HDIM
#define NQKV    1152        // 3*INNER
#define CPB_H   512
#define NTOK    (BATCH * NHW * NHW)          // 131072
#define OUT_ELEMS  (NTOK * DIMC)             // 50331648
#define QKV_ELEMS  (BATCH * HEADS * NWIN * M2 * HDIM) // 50331648

typedef __attribute__((ext_vector_type(16))) _Float16 v16h;
typedef __attribute__((ext_vector_type(8)))  float    v8f;
typedef __attribute__((ext_vector_type(4)))  unsigned v4u;
typedef __attribute__((ext_vector_type(8)))  int      v8i;
typedef __attribute__((ext_vector_type(4)))  int      v4i;

union F16Frag { v16h v; unsigned u[8]; };

// A-matrix fragment (16xK tile, row-major f16 in LDS, strideU32 = halves/2 per row).
// Contiguous k order (this code's choice, same for A and B): lane m=lane&15 holds row m; VGPR r holds K pair
// k0 = 2r + 16*(lane/16)  (halves) -> u32 index = k0/2.
__device__ __forceinline__ v16h load_a_frag(const unsigned* lds, int rowBase,
                                            int strideU32, int kBaseU32) {
  int lane = threadIdx.x & 31;
  int g = lane >> 4, m = lane & 15;
  F16Frag f;
#pragma unroll
  for (int r = 0; r < 8; ++r) {
    int kU = r + (g << 3);
    f.u[r] = lds[(rowBase + m) * strideU32 + kBaseU32 + kU];
  }
  return f.v;
}

// B-matrix fragment from [N][K]-major f16 LDS tile.
// Contiguous k order (this code's choice, same as A): lane n=lane&15 holds column n; VGPR r holds K pair
// k0 = 2r + 16*(lane/16) -> u32 index = r + 8*(lane/16).
__device__ __forceinline__ v16h load_b_frag(const unsigned* lds, int nBase,
                                            int strideU32, int kBaseU32) {
  int lane = threadIdx.x & 31;
  int g = lane >> 4, n = lane & 15;
  F16Frag f;
#pragma unroll
  for (int r = 0; r < 8; ++r)
    f.u[r] = lds[(nBase + n) * strideU32 + kBaseU32 + r + (g << 3)];
  return f.v;
}

__device__ __forceinline__ v8f wmma_f16(v16h a, v16h b, v8f c) {
  return __builtin_amdgcn_wmma_f32_16x16x32_f16(false, a, false, b, (short)0, c,
                                                false, false);
}

// ---------------- Tensor Data Mover: 2-D f16 tile -> LDS ----------------
// Loads tileRows rows of tileK halves (row stride = strideHalves halves) from
// gaddr into LDS at ldsOff, contiguous [tileRows][tileK] halves.
// D# group0: count=1 | lds_addr | global_addr(57b) | type=2.
// D# group1: data_size=1(2B), tensor_dim0/1, tile_dim0/1, tensor_dim0_stride.
__device__ __forceinline__ void tdm_load_tile_f16(unsigned ldsOff, const void* gaddr,
                                                  unsigned tileK, unsigned tileRows,
                                                  unsigned strideHalves) {
  unsigned long long ga = (unsigned long long)gaddr;
  v4u g0;
  g0[0] = 1u;                                   // count=1, user desc
  g0[1] = ldsOff;                               // lds_addr (bytes)
  g0[2] = (unsigned)ga;                         // global_addr[31:0]
  g0[3] = (unsigned)(ga >> 32) | (2u << 30);    // global_addr[56:32] | type=2
  v8i g1;
  g1[0] = (int)(1u << 16);                               // data_size=1 (2 bytes)
  g1[1] = (int)(tileK << 16);                            // tensor_dim0 lo16 (bits 63:48)
  g1[2] = (int)((tileK >> 16) | (tileRows << 16));       // tensor_dim0 hi | tensor_dim1 lo
  g1[3] = (int)((tileRows >> 16) | (tileK << 16));       // tensor_dim1 hi | tile_dim0
  g1[4] = (int)tileRows;                                 // tile_dim1 (tile_dim2=0)
  g1[5] = (int)strideHalves;                             // tensor_dim0_stride[31:0]
  g1[6] = 0;                                             // stride hi | dim1_stride lo
  g1[7] = 0;
  v4i z4 = {0, 0, 0, 0};
#if defined(__clang_major__) && __clang_major__ >= 23
  v8i z8 = {0, 0, 0, 0, 0, 0, 0, 0};
  __builtin_amdgcn_tensor_load_to_lds(g0, g1, z4, z4, z8, 0);
#else
  __builtin_amdgcn_tensor_load_to_lds(g0, g1, z4, z4, 0);
#endif
}

__device__ __forceinline__ unsigned lds_offset_of(const void* p) {
  return (unsigned)(unsigned long long)p;   // low 32 bits of LDS aperture address
}

// ---------------- kernel 1: f32 -> f16 weight convert ----------------
__global__ void cvt_f32_f16(const float* __restrict__ src, _Float16* __restrict__ dst, int n) {
  int i = blockIdx.x * 256 + threadIdx.x;
  if (i < n) dst[i] = (_Float16)src[i];
}

// ---------------- kernel 2: CPB-MLP relative position bias table ----------------
__global__ __launch_bounds__(256) void cpb_bias_kernel(
    const float* __restrict__ w1, const float* __restrict__ b1,
    const float* __restrict__ w2, const float* __restrict__ b2,
    float* __restrict__ bias) {
  int ij = blockIdx.x;                 // 0..4095
  int i = ij >> 6, j = ij & 63;
  float xi = (float)(i >> 3), yi = (float)(i & 7);
  float xj = (float)(j >> 3), yj = (float)(j & 7);
  float r0 = xj - xi, r1 = yj - yi;
  float c0 = (r0 >= 0.f ? 1.f : -1.f) * log1pf(fabsf(r0));
  float c1 = (r1 >= 0.f ? 1.f : -1.f) * log1pf(fabsf(r1));
  int t = threadIdx.x;
  float h0 = fmaxf(c0 * w1[2 * t]         + c1 * w1[2 * t + 1]         + b1[t],       0.f);
  float h1 = fmaxf(c0 * w1[2 * (t + 256)] + c1 * w1[2 * (t + 256) + 1] + b1[t + 256], 0.f);
  __shared__ float red[256];
  for (int h = 0; h < HEADS; ++h) {
    red[t] = h0 * w2[h * CPB_H + t] + h1 * w2[h * CPB_H + t + 256];
    __syncthreads();
    for (int s = 128; s > 0; s >>= 1) {
      if (t < s) red[t] += red[t + s];
      __syncthreads();
    }
    if (t == 0) bias[(h << 12) + ij] = red[0] + b2[h];
    __syncthreads();
  }
}

// ---------------- kernel 3: shifted QKV projection GEMM ----------------
// Block tile M=128, N=64; 4 waves, each wave 32x64 (8 WMMA / K-step).
// B (64x384 f16 weight tile) loaded ONCE per block by the TDM; A staged per
// K-step with f32->f16 conversion.
__global__ __launch_bounds__(128) void qkv_gemm(
    const float* __restrict__ x, const _Float16* __restrict__ wh,
    _Float16* __restrict__ qh, _Float16* __restrict__ kh, _Float16* __restrict__ vh) {
  __shared__ unsigned As[128 * 16];      // 128 rows x 32 halves (8 KB)
  __shared__ unsigned Bs[64 * 192];      // 64 N-rows x 384 K halves (48 KB)
  _Float16* Ah = (_Float16*)As;
  const int t = threadIdx.x;
  const int wv = t >> 5;
  const int lane = t & 31, g = lane >> 4, nl = lane & 15;
  const int mBlock = blockIdx.y * 128;
  const int nBlock = blockIdx.x * 64;

  if (wv == 0) {   // wave-uniform: only wave 0 issues the tensor DMA
    tdm_load_tile_f16(lds_offset_of(Bs), wh + nBlock * DIMC, DIMC, 64, DIMC);
    __builtin_amdgcn_s_wait_tensorcnt(0);
  }

  const v8f z8 = {0.f, 0.f, 0.f, 0.f, 0.f, 0.f, 0.f, 0.f};
  v8f acc[2][4] = {{z8, z8, z8, z8}, {z8, z8, z8, z8}};

  // A source row for this thread (one row each), with roll(-4,-4)
  int m = mBlock + t;
  int b = m >> 12, rem = m & 4095;
  int h = rem >> 6, w = rem & 63;
  int hs = (h + 4) & 63, ws = (w + 4) & 63;
  const float* xrow = x + (((b << 6) + hs) * NHW + ws) * DIMC;

  __syncthreads();   // TDM complete (wave0 waited) before anyone reads Bs

  for (int k0 = 0; k0 < DIMC; k0 += 32) {
    const float* srcA = xrow + k0;
    if (k0 + 32 < DIMC) __builtin_prefetch(srcA + 32, 0, 0);
#pragma unroll
    for (int i = 0; i < 32; ++i)
      Ah[t * 32 + i] = (_Float16)srcA[i];
    __syncthreads();
    v16h a0 = load_a_frag(As, wv * 32,      16, 0);
    v16h a1 = load_a_frag(As, wv * 32 + 16, 16, 0);
#pragma unroll
    for (int nt = 0; nt < 4; ++nt) {
      v16h bF = load_b_frag(Bs, nt * 16, 192, k0 >> 1);
      acc[0][nt] = wmma_f16(a0, bF, acc[0][nt]);
      acc[1][nt] = wmma_f16(a1, bF, acc[1][nt]);
    }
    __syncthreads();
  }
  // scatter into windowed q/k/v layout
#pragma unroll
  for (int mt = 0; mt < 2; ++mt) {
#pragma unroll
    for (int nt = 0; nt < 4; ++nt) {
#pragma unroll
      for (int r = 0; r < 8; ++r) {
        int mm = mBlock + wv * 32 + mt * 16 + r + (g << 3);
        int o  = nBlock + nt * 16 + nl;
        int qkvIdx = o / INNER;
        int orem = o - qkvIdx * INNER;
        int head = orem >> 5, d = orem & 31;
        int bb = mm >> 12, mr = mm & 4095;
        int hh = mr >> 6, ww = mr & 63;
        int win = ((hh >> 3) << 3) + (ww >> 3);
        int m2i = ((hh & 7) << 3) + (ww & 7);
        int idx = (((bb * HEADS + head) * NWIN + win) * M2 + m2i) * HDIM + d;
        _Float16 hv = (_Float16)acc[mt][nt][r];
        if (qkvIdx == 0) qh[idx] = hv;
        else if (qkvIdx == 1) kh[idx] = hv;
        else vh[idx] = hv;
      }
    }
  }
}

// ---------------- kernel 4: fused cosine window attention ----------------
// One block (4 waves) per (win, head, b).  q/k tiles arrive via TDM; v is
// transposed manually into [d][j].  S = q_n k_n^T/tau + bias + mask, softmax,
// attn -> d_out (f32), O = P V -> merged f16 [tok][INNER].
__global__ __launch_bounds__(128) void win_attn(
    const _Float16* __restrict__ qh, const _Float16* __restrict__ kh,
    const _Float16* __restrict__ vh, const float* __restrict__ bias,
    const float* __restrict__ tau, float* __restrict__ attn_out,
    _Float16* __restrict__ oh) {
  __shared__ unsigned qs[64 * 16];   // q  [i][d] 64x32 halves
  __shared__ unsigned ks[64 * 16];   // k  [j][d] 64x32 halves ([N][K] for S)
  __shared__ unsigned vt[32 * 32];   // v^T[d][j] 32x64 halves ([N][K] for O)
  __shared__ unsigned ps[64 * 32];   // P  [i][j] 64x64 halves (A for O)
  __shared__ float qscale[64], kscale[64];
  _Float16* qsh = (_Float16*)qs;
  _Float16* ksh = (_Float16*)ks;
  _Float16* vth = (_Float16*)vt;
  _Float16* psh = (_Float16*)ps;

  const int win = blockIdx.x, head = blockIdx.y, b = blockIdx.z;
  const int wy = win >> 3, wx = win & 7;
  const int base = (((b * HEADS + head) * NWIN + win) * M2) * HDIM;
  const int t = threadIdx.x, wv = t >> 5;
  const int lane = t & 31, g = lane >> 4, nl = lane & 15;

  if (wv == 0) {   // TDM: q and k tiles, 1-D linear 2048 halves each
    tdm_load_tile_f16(lds_offset_of(qs), qh + base, M2 * HDIM, 1, M2 * HDIM);
    tdm_load_tile_f16(lds_offset_of(ks), kh + base, M2 * HDIM, 1, M2 * HDIM);
  }
  for (int e = t; e < M2 * HDIM; e += 128) {     // v -> v^T (manual)
    int j = e >> 5, d = e & 31;
    vth[d * 64 + j] = vh[base + e];
  }
  if (wv == 0) __builtin_amdgcn_s_wait_tensorcnt(0);
  __syncthreads();

  // row norms (q gets 1/max(tau,.01) folded in)
  if (t < 64) {
    float ss = 0.f;
#pragma unroll
    for (int d = 0; d < HDIM; ++d) { float f = (float)qsh[t * HDIM + d]; ss += f * f; }
    float invtau = 1.0f / fmaxf(tau[head], 0.01f);
    qscale[t] = invtau / fmaxf(sqrtf(ss), 1e-12f);
  } else {
    int r = t - 64;
    float ss = 0.f;
#pragma unroll
    for (int d = 0; d < HDIM; ++d) { float f = (float)ksh[r * HDIM + d]; ss += f * f; }
    kscale[r] = 1.0f / fmaxf(sqrtf(ss), 1e-12f);
  }
  __syncthreads();
  for (int e = t; e < M2 * HDIM; e += 128) {
    int r = e >> 5;
    qsh[e] = (_Float16)((float)qsh[e] * qscale[r]);
    ksh[e] = (_Float16)((float)ksh[e] * kscale[r]);
  }
  __syncthreads();

  // S = q k^T : wave wv -> rows [16wv,16wv+16), 4 N tiles, K=32 single shot
  const v8f z8 = {0.f, 0.f, 0.f, 0.f, 0.f, 0.f, 0.f, 0.f};
  v8f acc[4] = {z8, z8, z8, z8};
  {
    v16h a = load_a_frag(qs, wv * 16, 16, 0);
#pragma unroll
    for (int nt = 0; nt < 4; ++nt) {
      v16h bF = load_b_frag(ks, nt * 16, 16, 0);
      acc[nt] = wmma_f16(a, bF, acc[nt]);
    }
  }

  // bias + shift masks + softmax (rows live in one 16-lane half)
  const float* bslice = bias + head * (M2 * M2);
  const int aBase0 = (((b * HEADS + head) * NWIN + win) * M2) * M2;
#pragma unroll
  for (int r = 0; r < 8; ++r) {
    int i = wv * 16 + r + (g << 3);
    float sv[4];
    float mx = -1e30f;
#pragma unroll
    for (int nt = 0; nt < 4; ++nt) {
      int j = nt * 16 + nl;
      float s = acc[nt][r] + bslice[i * M2 + j];
      if (wy == 7 && ((i >= 32) != (j >= 32))) s = -1e30f;                  // UL mask
      if (wx == 7 && (((i & 7) >= 4) != ((j & 7) >= 4))) s = -1e30f;        // LR mask
      sv[nt] = s;
      mx = fmaxf(mx, s);
    }
#pragma unroll
    for (int off = 8; off >= 1; off >>= 1) mx = fmaxf(mx, __shfl_xor(mx, off, 32));
    float sum = 0.f;
#pragma unroll
    for (int nt = 0; nt < 4; ++nt) { sv[nt] = __expf(sv[nt] - mx); sum += sv[nt]; }
#pragma unroll
    for (int off = 8; off >= 1; off >>= 1) sum += __shfl_xor(sum, off, 32);
    float inv = 1.0f / sum;
#pragma unroll
    for (int nt = 0; nt < 4; ++nt) {
      int j = nt * 16 + nl;
      float p = sv[nt] * inv;
      attn_out[aBase0 + i * M2 + j] = p;       // attn output (f32)
      psh[i * 64 + j] = (_Float16)p;           // P for O = P V
    }
  }
  __syncthreads();

  // O = P V : 2 N tiles (head_dim 32), K loop of 2 chunks of 32
  v8f oacc[2] = {z8, z8};
#pragma unroll
  for (int kc = 0; kc < 2; ++kc) {
    v16h pa = load_a_frag(ps, wv * 16, 32, kc * 16);
#pragma unroll
    for (int nt = 0; nt < 2; ++nt) {
      v16h vb = load_b_frag(vt, nt * 16, 32, kc * 16);
      oacc[nt] = wmma_f16(pa, vb, oacc[nt]);
    }
  }
  // merge windows -> f16 [tok][INNER]
#pragma unroll
  for (int nt = 0; nt < 2; ++nt) {
#pragma unroll
    for (int r = 0; r < 8; ++r) {
      int i = wv * 16 + r + (g << 3);
      int d = nt * 16 + nl;
      int hh = wy * 8 + (i >> 3), ww = wx * 8 + (i & 7);
      int tok = ((b << 6) + hh) * NHW + ww;
      oh[tok * INNER + head * HDIM + d] = (_Float16)oacc[nt][r];
    }
  }
}

// ---------------- kernel 5: output projection + inverse roll ----------------
// Same tiling as qkv_gemm: TDM-resident 64x384 B tile, per-K-step f16 A copy.
__global__ __launch_bounds__(128) void out_gemm(
    const _Float16* __restrict__ ao, const _Float16* __restrict__ wh,
    const float* __restrict__ b_out, float* __restrict__ out) {
  __shared__ unsigned As[128 * 16];
  __shared__ unsigned Bs[64 * 192];
  const int t = threadIdx.x;
  const int wv = t >> 5;
  const int lane = t & 31, g = lane >> 4, nl = lane & 15;
  const int mBlock = blockIdx.y * 128;
  const int nBlock = blockIdx.x * 64;

  if (wv == 0) {
    tdm_load_tile_f16(lds_offset_of(Bs), wh + nBlock * INNER, INNER, 64, INNER);
    __builtin_amdgcn_s_wait_tensorcnt(0);
  }

  const v8f z8 = {0.f, 0.f, 0.f, 0.f, 0.f, 0.f, 0.f, 0.f};
  v8f acc[2][4] = {{z8, z8, z8, z8}, {z8, z8, z8, z8}};
  const unsigned* aoRow = (const unsigned*)(ao + (mBlock + t) * INNER);

  __syncthreads();

  for (int k0 = 0; k0 < INNER; k0 += 32) {
    const unsigned* srcA = aoRow + (k0 >> 1);
#pragma unroll
    for (int i = 0; i < 16; ++i)
      As[t * 16 + i] = srcA[i];
    __syncthreads();
    v16h a0 = load_a_frag(As, wv * 32,      16, 0);
    v16h a1 = load_a_frag(As, wv * 32 + 16, 16, 0);
#pragma unroll
    for (int nt = 0; nt < 4; ++nt) {
      v16h bF = load_b_frag(Bs, nt * 16, 192, k0 >> 1);
      acc[0][nt] = wmma_f16(a0, bF, acc[0][nt]);
      acc[1][nt] = wmma_f16(a1, bF, acc[1][nt]);
    }
    __syncthreads();
  }
#pragma unroll
  for (int mt = 0; mt < 2; ++mt) {
#pragma unroll
    for (int nt = 0; nt < 4; ++nt) {
#pragma unroll
      for (int r = 0; r < 8; ++r) {
        int m = mBlock + wv * 32 + mt * 16 + r + (g << 3);
        int o = nBlock + nt * 16 + nl;
        int b = m >> 12, rem = m & 4095;
        int h = rem >> 6, w = rem & 63;
        int ho = (h + 4) & 63, wo = (w + 4) & 63;           // roll(+4,+4)
        out[(((b << 6) + ho) * NHW + wo) * DIMC + o] = acc[mt][nt][r] + b_out[o];
      }
    }
  }
}

// ---------------- launch ----------------
extern "C" void kernel_launch(void* const* d_in, const int* in_sizes, int n_in,
                              void* d_out, int out_size, void* d_ws, size_t ws_size,
                              hipStream_t stream) {
  const float* x     = (const float*)d_in[0];
  const float* w_qkv = (const float*)d_in[1];
  const float* w1    = (const float*)d_in[2];
  const float* b1    = (const float*)d_in[3];
  const float* w2    = (const float*)d_in[4];
  const float* b2    = (const float*)d_in[5];
  const float* tau   = (const float*)d_in[6];
  const float* w_out = (const float*)d_in[7];
  const float* b_out = (const float*)d_in[8];

  float* out_ptr  = (float*)d_out;                 // (32,64,64,384)
  float* attn_ptr = out_ptr + OUT_ELEMS;           // (32,12,64,64,64)

  // workspace carve-up
  _Float16* wqkv_h = (_Float16*)d_ws;              // 1152*384
  _Float16* wout_h = wqkv_h + NQKV * DIMC;         // 384*384
  float*    bias   = (float*)(wout_h + DIMC * INNER);  // 12*64*64
  _Float16* qh     = (_Float16*)(bias + HEADS * M2 * M2);
  _Float16* kh     = qh + QKV_ELEMS;
  _Float16* vh     = kh + QKV_ELEMS;
  _Float16* oh     = vh + QKV_ELEMS;               // merged attn output [tok][INNER]

  cvt_f32_f16<<<(NQKV * DIMC + 255) / 256, 256, 0, stream>>>(w_qkv, wqkv_h, NQKV * DIMC);
  cvt_f32_f16<<<(DIMC * INNER + 255) / 256, 256, 0, stream>>>(w_out, wout_h, DIMC * INNER);
  cpb_bias_kernel<<<M2 * M2, 256, 0, stream>>>(w1, b1, w2, b2, bias);
  qkv_gemm<<<dim3(NQKV / 64, NTOK / 128), 128, 0, stream>>>(x, wqkv_h, qh, kh, vh);
  win_attn<<<dim3(NWIN, HEADS, BATCH), 128, 0, stream>>>(qh, kh, vh, bias, tau, attn_ptr, oh);
  out_gemm<<<dim3(DIMC / 64, NTOK / 128), 128, 0, stream>>>(oh, wout_h, b_out, out_ptr);
}
